// AttDecoder_3874060501767
// MI455X (gfx1250) — matrix operand order repaired and checked by reading, not compiled, not run on hardware
//
#include <hip/hip_runtime.h>
#include <math.h>

// ---------------- problem constants ----------------
#define Bz 8
#define Tz 48
#define Cz 684
#define Hz 16
#define Wz 64
#define HIDz 256
#define INz 256
#define Az 512
#define Vz 111
#define LOCz 432
#define RATz 16
#define KCOVz 11
#define Pz (Hz*Wz)        // 1024 positions per batch row
#define BPz (Bz*Pz)       // 8192
#define KCz 128           // 121 (11x11) padded to 128 (4 x K32)
#define CKz 704           // 684 padded to 704 (22 x K32)
#define APITCH 130        // LDS A-tile pitch in ushorts (260B -> bank-conflict-free)

// output offsets (flat concat: word_probs, word_alphas, ctxs, outs)
#define O_PROBS  0
#define O_ALPHAS (Bz*Tz*Vz)                 // 42624
#define O_CTXS   (O_ALPHAS + Bz*Tz*Pz)      // 435840
#define O_OUTS   (O_CTXS  + Tz*Bz*Cz)       // 698496

typedef __attribute__((ext_vector_type(16))) __bf16          v16bf;
typedef __attribute__((ext_vector_type(16))) unsigned short  v16u;
typedef __attribute__((ext_vector_type(8)))  float           v8f;
typedef __attribute__((ext_vector_type(4)))  unsigned int    v4u;
typedef __attribute__((ext_vector_type(8)))  int             v8i;
typedef __attribute__((ext_vector_type(4)))  int             v4i;

#if __has_builtin(__builtin_amdgcn_tensor_load_to_lds)
#define HAVE_TDM 1
#else
#define HAVE_TDM 0
#endif

// ---------------- helpers ----------------
__device__ __forceinline__ unsigned short f2bf(float f) {
  unsigned int u = __float_as_uint(f);
  u = (u + 0x7FFFu + ((u >> 16) & 1u)) >> 16;   // RNE
  return (unsigned short)u;
}
__device__ __forceinline__ float sigmoidf_(float x) {
  return 1.0f / (1.0f + __expf(-x));
}
// K-index mapping of A fragment halves (16x32 bf16, ISA layout):
// lanes 0-15 -> M rows (K 0-7,16-23); lanes 16-31 -> same rows (K 8-15,24-31)
__device__ __forceinline__ int a_kidx(int h, int g) {
  return (h < 8) ? (g * 8 + h) : (16 + g * 8 + (h - 8));
}
// B fragment (32x16 bf16): lane%16 -> N col, element h -> K row a_kidx(h, lane/16). base=&B[k0][n0].
__device__ __forceinline__ v16bf load_B_bf(const unsigned short* base, int ld, int lane) {
  const unsigned short* col = base + (lane & 15);
  v16u u;
#pragma unroll
  for (int h = 0; h < 16; ++h) u[h] = col[a_kidx(h, lane >> 4) * ld];
  return __builtin_bit_cast(v16bf, u);
}

// ---------------- one-time prep kernels ----------------
__global__ void k_zero_f32(float* p, int n) {
  int i = blockIdx.x * blockDim.x + threadIdx.x;
  if (i < n) p[i] = 0.0f;
}

// mask2 = images_mask[:, :, ::16, ::16]; avg pool; hidden0 = tanh(avg @ W_init + b_init)
__global__ void k_prep_mask_h0(const float* __restrict__ F, const float* __restrict__ im,
                               const float* __restrict__ W_init, const float* __restrict__ b_init,
                               float* __restrict__ mask2, float* __restrict__ hidden) {
  __shared__ float s_m[Pz];
  __shared__ float s_avg[Cz];
  __shared__ float s_red[256];
  const int b = blockIdx.x, tid = threadIdx.x;
  float lsum = 0.0f;
  for (int p = tid; p < Pz; p += 256) {
    int h = p >> 6, w = p & 63;
    float m = im[(size_t)b * (256 * 1024) + h * (RATz * 1024) + w * RATz];
    s_m[p] = m; mask2[b * Pz + p] = m; lsum += m;
  }
  s_red[tid] = lsum; __syncthreads();
  for (int s = 128; s; s >>= 1) { if (tid < s) s_red[tid] += s_red[tid + s]; __syncthreads(); }
  const float msum = s_red[0];
  for (int c = tid; c < Cz; c += 256) {
    const float* Fr = F + ((size_t)b * Cz + c) * Pz;
    float s = 0.0f;
    for (int p = 0; p < Pz; ++p) s += Fr[p] * s_m[p];
    s_avg[c] = s / msum;
  }
  __syncthreads();
  float hsum = b_init[tid];
  for (int c = 0; c < Cz; ++c) hsum += s_avg[c] * W_init[c * HIDz + tid];
  hidden[b * HIDz + tid] = tanhf(hsum);
}

__global__ void k_prep_locw(const float* __restrict__ loc, const float* __restrict__ W_loc,
                            const float* __restrict__ b_loc, float* __restrict__ locw) {
  int idx = blockIdx.x * 256 + threadIdx.x;       // B*HID = 2048
  int b = idx / HIDz, j = idx % HIDz;
  float s = b_loc[j];
  for (int l = 0; l < LOCz; ++l) s += loc[b * LOCz + l] * W_loc[l * HIDz + j];
  locw[idx] = s;
}

// K_eff[k][a] = sum_c K_cov[c][k] * W_cov[c][a]   (k<121, zero-padded to 128), bf16
__global__ void k_prep_keff(const float* __restrict__ K_cov, const float* __restrict__ W_cov,
                            unsigned short* __restrict__ Keff) {
  int idx = blockIdx.x * 256 + threadIdx.x;       // KCz*Az
  int k = idx / Az, a = idx % Az;
  float s = 0.0f;
  if (k < KCOVz * KCOVz)
    for (int c = 0; c < 512; ++c) s += K_cov[c * (KCOVz * KCOVz) + k] * W_cov[c * Az + a];
  Keff[idx] = f2bf(s);
}

// embeddings gather: tokens = [1, labels[:, :-1]]
__global__ void k_prep_embs(const int* __restrict__ labels, const float* __restrict__ emb,
                            float* __restrict__ embs) {
  int idx = blockIdx.x * 256 + threadIdx.x;       // T*B*IN
  int t = idx / (Bz * INz), r = idx % (Bz * INz);
  int b = r / INz, i = r % INz;
  int tok = (t == 0) ? 1 : labels[b * Tz + (t - 1)];
  embs[idx] = emb[tok * INz + i];
}

// Afeat[pg][c] = bf16(F[b][c][p]) with K padded to 704
__global__ void k_prep_afeat(const float* __restrict__ F, unsigned short* __restrict__ Afeat) {
  size_t idx = (size_t)blockIdx.x * 256 + threadIdx.x;  // BP*CK
  int pg = (int)(idx / CKz), c = (int)(idx % CKz);
  int b = pg >> 10, p = pg & 1023;
  float v = (c < Cz) ? F[((size_t)b * Cz + c) * Pz + p] : 0.0f;
  Afeat[idx] = f2bf(v);
}
// Bfeat[c][a] = bf16(K_feat[a][c])
__global__ void k_prep_bfeat(const float* __restrict__ K_feat, unsigned short* __restrict__ Bfeat) {
  int idx = blockIdx.x * 256 + threadIdx.x;       // CK*A
  int c = idx / Az, a = idx % Az;
  float v = (c < Cz) ? K_feat[a * Cz + c] : 0.0f;
  Bfeat[idx] = f2bf(v);
}

// trans_p[pg][a] = Afeat(8192x704) @ Bfeat(704x512) + b_feat   via bf16 WMMA.
// The 16x704 A-stripe (22528 B, contiguous) is staged into LDS by the Tensor
// Data Mover (1-D descriptor), then all 4 waves read fragments from LDS.
__global__ void k_trans_p(const unsigned short* __restrict__ Afeat,
                          const unsigned short* __restrict__ Bfeat,
                          const float* __restrict__ b_feat, float* __restrict__ transp) {
  __shared__ unsigned short sA[16 * CKz];                     // 22528 B
  const int lane = threadIdx.x & 31, wv = threadIdx.x >> 5;   // 4 waves
  const int pg0 = blockIdx.x * 16;
  const int a0  = (blockIdx.y * 4 + wv) * 16;
#if HAVE_TDM
  if (threadIdx.x < 32) {                                     // wave 0 drives the TDM
    unsigned long long ga = (unsigned long long)(const void*)(Afeat + (size_t)pg0 * CKz);
    unsigned lds = (unsigned)(unsigned long long)(__attribute__((address_space(3))) unsigned short*)sA;
    const unsigned nel = (16u * CKz * 2u) / 8u;               // 2816 x 8-byte elements
    v4u g0 = {};
    g0[0] = 1u;                                               // count=1 (valid), load
    g0[1] = lds;                                              // lds_addr (bytes)
    g0[2] = (unsigned)(ga & 0xFFFFFFFFu);                     // global_addr[31:0]
    g0[3] = (unsigned)((ga >> 32) & 0x01FFFFFFu) | (2u << 30);// global_addr[56:32] | type=2
    v8i g1 = {};
    g1[0] = (int)(3u << 16);                                  // wg_mask=0, data_size=8B
    g1[1] = (int)((nel & 0xFFFFu) << 16);                     // tensor_dim0 lo16
    g1[2] = (int)((nel >> 16) & 0xFFFFu) | (1 << 16);         // tensor_dim0 hi16 | tensor_dim1=1
    g1[3] = (int)((nel & 0xFFFFu) << 16);                     // tile_dim0 = 2816
    g1[4] = 0;                                                // tile_dim1=0 (1-D), tile_dim2=0
    g1[5] = (int)nel;                                         // tensor_dim0_stride lo32
    g1[6] = (int)((nel & 0xFFFFu) << 16);                     // stride0 hi=0 | stride1 lo16
    g1[7] = (int)(nel >> 16);                                 // stride1 hi
    v4i g2 = {}; v4i g3 = {};
    v8i g4 = {};                                              // extra group (6-arg toolchain), zero-filled
    __builtin_amdgcn_tensor_load_to_lds(g0, g1, g2, g3, g4, 0);
    __builtin_amdgcn_s_wait_tensorcnt(0);
  }
  __syncthreads();
#else
  for (int i = threadIdx.x; i < 16 * CKz; i += 128) sA[i] = Afeat[(size_t)pg0 * CKz + i];
  __syncthreads();
#endif
  v8f acc = {};
  const int m = lane & 15, g = lane >> 4;
  for (int ks = 0; ks < CKz / 32; ++ks) {
    v16u u;
#pragma unroll
    for (int h = 0; h < 16; ++h) u[h] = sA[m * CKz + ks * 32 + a_kidx(h, g)];
    v16bf a = __builtin_bit_cast(v16bf, u);
    v16bf bm = load_B_bf(Bfeat + (size_t)(ks * 32) * Az + a0, Az, lane);
    acc = __builtin_amdgcn_wmma_f32_16x16x32_bf16(false, a, false, bm, (short)0, acc, false, false);
  }
  const int n = lane & 15, mh = (lane >> 4) * 8;
  const float bias = b_feat[a0 + n];
#pragma unroll
  for (int r = 0; r < 8; ++r)
    transp[(size_t)(pg0 + mh + r) * Az + a0 + n] = acc[r] + bias;
}

// ---------------- per-step kernels ----------------
// GRU + query, one block per batch row
__global__ void k_gru_query(const float* __restrict__ embs, float* __restrict__ hidden,
                            const float* __restrict__ W_ih, const float* __restrict__ W_hh,
                            const float* __restrict__ b_ih, const float* __restrict__ b_hh,
                            const float* __restrict__ W_q, const float* __restrict__ b_q,
                            float* __restrict__ query, int t) {
  __shared__ float sh[HIDz], se[INz], sn[HIDz];
  const int b = blockIdx.x, j = threadIdx.x;
  sh[j] = hidden[b * HIDz + j];
  se[j] = embs[((size_t)t * Bz + b) * INz + j];
  __syncthreads();
  float gir = 0, giz = 0, gin = 0, ghr = 0, ghz = 0, ghn = 0;
  const float* Wr  = W_ih + (size_t)j * INz;
  const float* Wz_ = W_ih + (size_t)(256 + j) * INz;
  const float* Wn  = W_ih + (size_t)(512 + j) * INz;
  const float* Ur  = W_hh + (size_t)j * HIDz;
  const float* Uz  = W_hh + (size_t)(256 + j) * HIDz;
  const float* Un  = W_hh + (size_t)(512 + j) * HIDz;
  for (int k = 0; k < HIDz; ++k) {
    float x = se[k], h = sh[k];
    gir += x * Wr[k]; giz += x * Wz_[k]; gin += x * Wn[k];
    ghr += h * Ur[k]; ghz += h * Uz[k];  ghn += h * Un[k];
  }
  float r = sigmoidf_(gir + b_ih[j]       + ghr + b_hh[j]);
  float z = sigmoidf_(giz + b_ih[256 + j] + ghz + b_hh[256 + j]);
  float n = tanhf(gin + b_ih[512 + j] + r * (ghn + b_hh[512 + j]));
  float hn = (1.0f - z) * n + z * sh[j];
  sn[j] = hn; hidden[b * HIDz + j] = hn;
  __syncthreads();
  for (int a = j; a < Az; a += 256) {
    float q = b_q[a];
    for (int k = 0; k < HIDz; ++k) q += sn[k] * W_q[(size_t)k * Az + a];
    query[b * Az + a] = q;
  }
}

// Fused: im2col(alpha_sum)->LDS (bf16, bank-padded) -> WMMA (16x128 @ 128x512)
//        + query + trans_p -> tanh -> dot w_alpha -> energy
__global__ void k_attn_energy(const float* __restrict__ asum,
                              const unsigned short* __restrict__ Keff,
                              const float* __restrict__ query, const float* __restrict__ transp,
                              const float* __restrict__ w_alpha, const float* __restrict__ b_alpha,
                              float* __restrict__ energy) {
  __shared__ unsigned short sA[16 * APITCH];    // 16 rows x 130 (pitch) = 4160 B
  __shared__ float part[8 * 16];
  const int lane = threadIdx.x & 31, wv = threadIdx.x >> 5;   // 8 waves
  const int pg0 = blockIdx.x * 16, b = pg0 >> 10, p0 = pg0 & 1023;
  // --- build the im2col A-tile directly in LDS (16 positions x 121-pt window) ---
  for (int e = threadIdx.x; e < 16 * KCz; e += 256) {
    const int i = e >> 7, kk = e & 127;
    float v = 0.0f;
    if (kk < KCOVz * KCOVz) {
      const int p = p0 + i;
      const int h = (p >> 6) + kk / KCOVz - KCOVz / 2;
      const int w = (p & 63) + kk % KCOVz - KCOVz / 2;
      if (h >= 0 && h < Hz && w >= 0 && w < Wz) v = asum[b * Pz + h * Wz + w];
    }
    sA[i * APITCH + kk] = f2bf(v);
  }
  __syncthreads();
  // --- preload the 4 A fragments once; reuse across all 4 N-tiles ---
  const int m = lane & 15, g = lane >> 4;
  v16bf Afr[4];
#pragma unroll
  for (int ks = 0; ks < 4; ++ks) {
    v16u u;
#pragma unroll
    for (int h = 0; h < 16; ++h) u[h] = sA[m * APITCH + ks * 32 + a_kidx(h, g)];
    Afr[ks] = __builtin_bit_cast(v16bf, u);
  }
  float s[8];
#pragma unroll
  for (int r = 0; r < 8; ++r) s[r] = 0.0f;
  for (int nt = 0; nt < 4; ++nt) {                            // 64 channels per wave
    const int a0 = wv * 64 + nt * 16;
    v8f acc = {};
#pragma unroll
    for (int ks = 0; ks < 4; ++ks) {
      v16bf bm = load_B_bf(Keff + (size_t)(ks * 32) * Az + a0, Az, lane);
      acc = __builtin_amdgcn_wmma_f32_16x16x32_bf16(false, Afr[ks], false, bm, (short)0, acc, false, false);
    }
    const int n = lane & 15, mh = (lane >> 4) * 8, a = a0 + n;
    const float q = query[b * Az + a], wa = w_alpha[a];
#pragma unroll
    for (int r = 0; r < 8; ++r) {
      float val = tanhf(acc[r] + q + transp[(size_t)(pg0 + mh + r) * Az + a]);
      s[r] += val * wa;
    }
  }
#pragma unroll
  for (int msk = 1; msk < 16; msk <<= 1)
#pragma unroll
    for (int r = 0; r < 8; ++r) s[r] += __shfl_xor(s[r], msk, 32);
  if (lane == 0)  for (int r = 0; r < 8; ++r) part[wv * 16 + r]     = s[r];
  if (lane == 16) for (int r = 0; r < 8; ++r) part[wv * 16 + 8 + r] = s[r];
  __syncthreads();
  if (threadIdx.x < 16) {
    float e = b_alpha[0];
    for (int w2 = 0; w2 < 8; ++w2) e += part[w2 * 16 + threadIdx.x];
    energy[pg0 + threadIdx.x] = e;
  }
}

// Fused softmax: global max, masked exp, per-batch segmented sums, normalize,
// alpha_sum update, thresholded alpha, alpha output. Single block, 1024 threads.
__global__ void k_softmax(float* __restrict__ energy, const float* __restrict__ mask2,
                          float* __restrict__ asum, float* __restrict__ alpham,
                          float* __restrict__ out, int t) {
  __shared__ float red[1024];
  const int tid = threadIdx.x;
  float mx = -3.4e38f;
  for (int i = tid; i < BPz; i += 1024) mx = fmaxf(mx, energy[i]);
  red[tid] = mx; __syncthreads();
  for (int s = 512; s; s >>= 1) { if (tid < s) red[tid] = fmaxf(red[tid], red[tid + s]); __syncthreads(); }
  const float gm = red[0];
  __syncthreads();
  const int b = tid >> 7, l = tid & 127;                      // 128 threads per batch row
  float lsum = 0.0f;
  for (int p = l; p < Pz; p += 128) {
    float e = __expf(energy[b * Pz + p] - gm) * mask2[b * Pz + p];
    energy[b * Pz + p] = e; lsum += e;
  }
  red[tid] = lsum; __syncthreads();
  for (int s = 64; s; s >>= 1) { if (l < s) red[tid] += red[tid + s]; __syncthreads(); }
  const float inv = 1.0f / (red[b << 7] + 1e-10f);
  for (int p = l; p < Pz; p += 128) {
    float a = energy[b * Pz + p] * inv;
    asum[b * Pz + p] += a;
    alpham[b * Pz + p] = (a > 0.02f) ? a : 0.0f;
    out[O_ALPHAS + ((size_t)b * Tz + t) * Pz + p] = a;
  }
}

// ctx[b][c] = sum_p alpham[b][p] * F[b][c][p]  (one wave per output)
__global__ void k_ctx(const float* __restrict__ alpham, const float* __restrict__ F,
                      float* __restrict__ ctx, float* __restrict__ out, int t) {
  const int lane = threadIdx.x & 31;
  const int gw = blockIdx.x * 8 + (threadIdx.x >> 5);
  if (gw >= Bz * Cz) return;
  const int b = gw / Cz, c = gw % Cz;
  const float* Fr = F + ((size_t)b * Cz + c) * Pz;
  const float* Ar = alpham + b * Pz;
  float s = 0.0f;
  for (int p = lane; p < Pz; p += 32) s += Ar[p] * Fr[p];
#pragma unroll
  for (int off = 16; off; off >>= 1) s += __shfl_xor(s, off, 32);
  if (lane == 0) { ctx[gw] = s; out[O_CTXS + ((size_t)t * Bz + b) * Cz + c] = s; }
}

// out_state = max(h@W_state + e@W_emb + ctx@W_ctx + biases, loc_w); prob = out_state@W_out + b_out
__global__ void k_out_prob(const float* __restrict__ hidden, const float* __restrict__ embs,
                           const float* __restrict__ ctx,
                           const float* __restrict__ W_state, const float* __restrict__ b_state,
                           const float* __restrict__ W_emb,   const float* __restrict__ b_emb,
                           const float* __restrict__ W_ctx,   const float* __restrict__ b_ctx,
                           const float* __restrict__ W_out,   const float* __restrict__ b_out,
                           const float* __restrict__ locw, float* __restrict__ out, int t) {
  __shared__ float sh[HIDz], se[INz], sc[Cz], so[HIDz];
  const int b = blockIdx.x, j = threadIdx.x;
  sh[j] = hidden[b * HIDz + j];
  se[j] = embs[((size_t)t * Bz + b) * INz + j];
  for (int c = j; c < Cz; c += 256) sc[c] = ctx[b * Cz + c];
  __syncthreads();
  float acc = b_state[j] + b_emb[j] + b_ctx[j];
  for (int k = 0; k < HIDz; ++k)
    acc += sh[k] * W_state[(size_t)k * HIDz + j] + se[k] * W_emb[(size_t)k * HIDz + j];
  for (int c = 0; c < Cz; ++c) acc += sc[c] * W_ctx[(size_t)c * HIDz + j];
  acc = fmaxf(acc, locw[b * HIDz + j]);
  so[j] = acc;
  out[O_OUTS + ((size_t)t * Bz + b) * HIDz + j] = acc;
  __syncthreads();
  if (j < Vz) {
    float p = b_out[j];
    for (int k = 0; k < HIDz; ++k) p += so[k] * W_out[(size_t)k * Vz + j];
    out[O_PROBS + ((size_t)b * Tz + t) * Vz + j] = p;
  }
}

// ---------------- host launcher ----------------
extern "C" void kernel_launch(void* const* d_in, const int* in_sizes, int n_in,
                              void* d_out, int out_size, void* d_ws, size_t ws_size,
                              hipStream_t stream) {
  (void)in_sizes; (void)n_in; (void)out_size; (void)ws_size;
  const float* F       = (const float*)d_in[0];
  const int*   labels  = (const int*)  d_in[1];
  const float* locp    = (const float*)d_in[2];
  const float* im      = (const float*)d_in[3];
  // d_in[4] labels_mask: unused by reference outputs
  const float* W_init  = (const float*)d_in[5];
  const float* b_init  = (const float*)d_in[6];
  const float* emb     = (const float*)d_in[7];
  const float* W_ih    = (const float*)d_in[8];
  const float* W_hh    = (const float*)d_in[9];
  const float* b_ih    = (const float*)d_in[10];
  const float* b_hh    = (const float*)d_in[11];
  const float* W_q     = (const float*)d_in[12];
  const float* b_q     = (const float*)d_in[13];
  const float* K_cov   = (const float*)d_in[14];
  const float* W_cov   = (const float*)d_in[15];
  const float* w_alpha = (const float*)d_in[16];
  const float* b_alpha = (const float*)d_in[17];
  const float* K_feat  = (const float*)d_in[18];
  const float* b_feat  = (const float*)d_in[19];
  const float* W_state = (const float*)d_in[20];
  const float* b_state = (const float*)d_in[21];
  const float* W_emb   = (const float*)d_in[22];
  const float* b_emb   = (const float*)d_in[23];
  const float* W_ctx   = (const float*)d_in[24];
  const float* b_ctx   = (const float*)d_in[25];
  const float* W_out   = (const float*)d_in[26];
  const float* b_out   = (const float*)d_in[27];
  const float* W_loc   = (const float*)d_in[28];
  const float* b_loc   = (const float*)d_in[29];
  float* out = (float*)d_out;

  // workspace carve-up
  char* base = (char*)d_ws;
  size_t off = 0;
  auto carve = [&](size_t bytes) { char* p = base + off; off = (off + bytes + 255) & ~(size_t)255; return p; };
  float* hidden = (float*)carve(Bz * HIDz * 4);
  float* mask2  = (float*)carve(BPz * 4);
  float* asum   = (float*)carve(BPz * 4);
  float* alpham = (float*)carve(BPz * 4);
  float* energy = (float*)carve(BPz * 4);
  float* locw   = (float*)carve(Bz * HIDz * 4);
  float* query  = (float*)carve(Bz * Az * 4);
  float* embs   = (float*)carve((size_t)Tz * Bz * INz * 4);
  float* ctx    = (float*)carve(Bz * Cz * 4);
  float* transp = (float*)carve((size_t)BPz * Az * 4);
  unsigned short* Keff  = (unsigned short*)carve((size_t)KCz * Az * 2);
  unsigned short* Afeat = (unsigned short*)carve((size_t)BPz * CKz * 2);
  unsigned short* Bfeat = (unsigned short*)carve((size_t)CKz * Az * 2);

  // ---- one-time precompute ----
  k_zero_f32    <<<BPz / 256, 256, 0, stream>>>(asum, BPz);
  k_prep_mask_h0<<<Bz, 256, 0, stream>>>(F, im, W_init, b_init, mask2, hidden);
  k_prep_locw   <<<(Bz * HIDz) / 256, 256, 0, stream>>>(locp, W_loc, b_loc, locw);
  k_prep_keff   <<<(KCz * Az) / 256, 256, 0, stream>>>(K_cov, W_cov, Keff);
  k_prep_embs   <<<(Tz * Bz * INz) / 256, 256, 0, stream>>>(labels, emb, embs);
  k_prep_afeat  <<<(BPz * CKz) / 256, 256, 0, stream>>>(F, Afeat);
  k_prep_bfeat  <<<(CKz * Az) / 256, 256, 0, stream>>>(K_feat, Bfeat);
  k_trans_p     <<<dim3(BPz / 16, Az / 64), 128, 0, stream>>>(Afeat, Bfeat, b_feat, transp);

  // ---- sequential scan (48 steps, 5 kernels/step; all graph-captured) ----
  for (int t = 0; t < Tz; ++t) {
    k_gru_query  <<<Bz, 256, 0, stream>>>(embs, hidden, W_ih, W_hh, b_ih, b_hh, W_q, b_q, query, t);
    k_attn_energy<<<BPz / 16, 256, 0, stream>>>(asum, Keff, query, transp, w_alpha, b_alpha, energy);
    k_softmax    <<<1, 1024, 0, stream>>>(energy, mask2, asum, alpham, out, t);
    k_ctx        <<<(Bz * Cz + 7) / 8, 256, 0, stream>>>(alpham, F, ctx, out, t);
    k_out_prob   <<<Bz, 256, 0, stream>>>(hidden, embs, ctx, W_state, b_state, W_emb, b_emb,
                                          W_ctx, b_ctx, W_out, b_out, locw, out, t);
  }
}
